// TrendRefinement_40011915329799
// MI455X (gfx1250) — matrix operand order repaired and checked by reading, not compiled, not run on hardware
//
#include <hip/hip_runtime.h>

// ---------------------------------------------------------------------------
// TrendRefinement for MI455X (gfx1250, wave32, WMMA, async global->LDS).
//   t   = inv ⊙ (U^T V)   per (b,h)   [512x512 x 512x64 GEMM, f16 WMMA]
//   out = U t              per (b,h)   [512x512 x 512x64 GEMM, f16 WMMA]
// All GEMM operands pre-transposed/converted so hot-loop staging is pure
// contiguous 16B copies (async-to-LDS), double-buffered in LDS.
// ---------------------------------------------------------------------------

typedef __attribute__((ext_vector_type(16))) _Float16 v16h;
typedef __attribute__((ext_vector_type(8)))  float    v8f;
typedef __attribute__((ext_vector_type(2)))  _Float16 h2;

#define SEQ_L   512
#define NHEADS  8
#define EDIM    64
#define NBATCH  16
#define BH      (NBATCH * NHEADS)
#define KTILE   32
#define MTILE   128
#define NKT     (SEQ_L / KTILE)   // 16
#define LDSS    40                // padded LDS row stride in halfs (80B)

#if __has_builtin(__builtin_amdgcn_global_load_async_to_lds_b128)
#define HAVE_ASYNC 1
#else
#define HAVE_ASYNC 0
#endif

// Pointee type the async builtin expects: int __attribute__((vector_size(16)))
typedef int v4i __attribute__((vector_size(16)));
typedef __attribute__((address_space(1))) v4i as1_v4i;
typedef __attribute__((address_space(3))) v4i as3_v4i;

// 16-byte global -> LDS copy (async DMA path if the toolchain exposes it).
__device__ __forceinline__ void cp16_lds(_Float16* l, const _Float16* g) {
#if HAVE_ASYNC
    __builtin_amdgcn_global_load_async_to_lds_b128((as1_v4i*)g, (as3_v4i*)l, 0, 0);
#else
    *(uint4*)l = *(const uint4*)g;
#endif
}

template <int N>
__device__ __forceinline__ void wait_async() {
#if HAVE_ASYNC
#if __has_builtin(__builtin_amdgcn_s_wait_asynccnt)
    __builtin_amdgcn_s_wait_asynccnt(N);
#else
    asm volatile("s_wait_asynccnt %0" ::"i"(N) : "memory");
#endif
#endif
}

// ---------------------------------------------------------------------------
// Prep kernel: U fp32 -> U16 (f16, same layout) and UT16 (f16, transposed).
// ---------------------------------------------------------------------------
__global__ __launch_bounds__(256) void prep_u_kernel(const float* __restrict__ U,
                                                     _Float16* __restrict__ U16,
                                                     _Float16* __restrict__ UT16) {
    __shared__ _Float16 tile[64][66];
    const int jt = blockIdx.x & 7, mt = blockIdx.x >> 3;
    const int tid = threadIdx.x;
    #pragma unroll
    for (int rep = 0; rep < 16; ++rep) {
        int lin = rep * 256 + tid;
        int jj = lin >> 6, mm = lin & 63;
        float v = U[(size_t)(jt * 64 + jj) * SEQ_L + mt * 64 + mm];
        _Float16 hv = (_Float16)v;
        U16[(size_t)(jt * 64 + jj) * SEQ_L + mt * 64 + mm] = hv;
        tile[jj][mm] = hv;
    }
    __syncthreads();
    #pragma unroll
    for (int rep = 0; rep < 16; ++rep) {
        int lin = rep * 256 + tid;
        int mm = lin >> 6, jj = lin & 63;
        UT16[(size_t)(mt * 64 + mm) * SEQ_L + jt * 64 + jj] = tile[jj][mm];
    }
}

// ---------------------------------------------------------------------------
// Prep kernel: values fp32 (B,L,H,E) -> vT16 f16 [bh][d][j]  (j contiguous)
// ---------------------------------------------------------------------------
__global__ __launch_bounds__(256) void prep_v_kernel(const float* __restrict__ values,
                                                     _Float16* __restrict__ vT16) {
    __shared__ _Float16 tile[64][66];
    const int j0 = blockIdx.x * 64;
    const int bh = blockIdx.y;
    const int b = bh >> 3, h = bh & 7;
    const int tid = threadIdx.x;
    #pragma unroll
    for (int rep = 0; rep < 16; ++rep) {
        int lin = rep * 256 + tid;
        int jj = lin >> 6, dd = lin & 63;
        float v = values[((size_t)(b * SEQ_L + j0 + jj) * NHEADS + h) * EDIM + dd];
        tile[jj][dd] = (_Float16)v;
    }
    __syncthreads();
    #pragma unroll
    for (int rep = 0; rep < 16; ++rep) {
        int lin = rep * 256 + tid;
        int dd = lin >> 6, jj = lin & 63;
        vT16[((size_t)(bh * EDIM + dd)) * SEQ_L + j0 + jj] = tile[jj][dd];
    }
}

// ---------------------------------------------------------------------------
// conv1d (replicate pad) + 1+elu + inv = 1/(1+lam*S).  inv layout [bh][l].
// ---------------------------------------------------------------------------
__global__ __launch_bounds__(256) void lambda_inv_kernel(const float* __restrict__ values,
                                                         const float* __restrict__ w,
                                                         const float* __restrict__ bias,
                                                         const float* __restrict__ S,
                                                         float* __restrict__ inv) {
    int idx = blockIdx.x * blockDim.x + threadIdx.x;   // [bh][l]
    int l  = idx & (SEQ_L - 1);
    int bh = idx >> 9;
    int h  = bh & (NHEADS - 1);
    int b  = bh >> 3;

    float y = bias[0];
    #pragma unroll
    for (int k = 0; k < 7; ++k) {
        int j = l + k - 3;
        j = j < 0 ? 0 : (j > SEQ_L - 1 ? SEQ_L - 1 : j);
        const float* vp = values + ((size_t)(b * SEQ_L + j) * NHEADS + h) * EDIM;
        for (int e = 0; e < EDIM; ++e)
            y = fmaf(w[e * 7 + k], vp[e], y);
    }
    float elu = y > 0.0f ? y : (__expf(y) - 1.0f);
    float lam = 1.0f + elu;
    inv[idx] = 1.0f / (1.0f + lam * S[l]);
}

// ---------------------------------------------------------------------------
// Fragment builders (contiguous k = 16*hi + s in both A and B, wave32) — contiguous dword
// LDS reads from [row][k] tiles (compile to ds_load_b128 pairs).
// ---------------------------------------------------------------------------
__device__ __forceinline__ v16h build_a_frag(const _Float16 (*Ast)[LDSS], int row, int hi) {
    v16h a;
    #pragma unroll
    for (int g = 0; g < 2; ++g)
        #pragma unroll
        for (int p = 0; p < 4; ++p) {
            h2 t = *(const h2*)&Ast[row][hi * 16 + g * 8 + p * 2];
            a[g * 8 + p * 2]     = t.x;
            a[g * 8 + p * 2 + 1] = t.y;
        }
    return a;
}

__device__ __forceinline__ v16h build_b_frag(const _Float16 (*Bt)[LDSS], int n, int hi) {
    v16h bfr;
    #pragma unroll
    for (int p = 0; p < 8; ++p) {
        h2 t = *(const h2*)&Bt[n][hi * 16 + p * 2];
        bfr[p * 2]     = t.x;
        bfr[p * 2 + 1] = t.y;
    }
    return bfr;
}

// Stage one K-tile: A 128x32 halfs, B 64x32 halfs, all contiguous 16B copies.
// 6 copy instructions per thread => per-wave ASYNCcnt += 6 per stage.
__device__ __forceinline__ void stage_tiles(_Float16 (*Ast)[LDSS], _Float16 (*Bt)[LDSS],
                                            const _Float16* __restrict__ Abase,
                                            const _Float16* __restrict__ Bbase, int tid) {
    #pragma unroll
    for (int i = 0; i < 4; ++i) {
        int idx = tid + i * 128;              // 0..511
        int m = idx >> 2, kq = idx & 3;
        cp16_lds(&Ast[m][kq * 8], Abase + (size_t)m * SEQ_L + kq * 8);
    }
    #pragma unroll
    for (int i = 0; i < 2; ++i) {
        int idx = tid + i * 128;              // 0..255
        int n = idx >> 2, kq = idx & 3;
        cp16_lds(&Bt[n][kq * 8], Bbase + (size_t)n * SEQ_L + kq * 8);
    }
}

// ---------------------------------------------------------------------------
// Unified GEMM: C[m][n] = sum_k Asrc[m0+m][k] * Bsrc[bh*64+n][k]
// PASS 1: Asrc=UT16, Bsrc=vT16, epilogue scales by inv, stores tT16 [bh][d][m]
// PASS 2: Asrc=U16,  Bsrc=tT16, epilogue stores fp32 out (B,L,H,E)
// Double-buffered LDS; staging of tile k+1 overlaps WMMA of tile k.
// ---------------------------------------------------------------------------
template <int PASS>
__global__ __launch_bounds__(128) void gemm_kernel(const _Float16* __restrict__ Asrc,
                                                   const _Float16* __restrict__ Bsrc,
                                                   const float* __restrict__ inv,
                                                   void* __restrict__ dstp) {
    __shared__ __align__(16) _Float16 Ast[2][MTILE][LDSS];
    __shared__ __align__(16) _Float16 Bt[2][EDIM][LDSS];

    const int m0   = blockIdx.x * MTILE;
    const int bh   = blockIdx.y;
    const int tid  = threadIdx.x;
    const int lane = tid & 31, wave = tid >> 5;
    const int r    = lane & 15, hi = lane >> 4;

    const _Float16* Abase = Asrc + (size_t)m0 * SEQ_L;
    const _Float16* Bbase = Bsrc + (size_t)bh * EDIM * SEQ_L;

    v8f acc[2][4] = {};

    stage_tiles(Ast[0], Bt[0], Abase, Bbase, tid);

    for (int it = 0; it < NKT; ++it) {
        const int cur = it & 1;
        if (it + 1 < NKT) {
            stage_tiles(Ast[cur ^ 1], Bt[cur ^ 1],
                        Abase + (it + 1) * KTILE, Bbase + (it + 1) * KTILE, tid);
            wait_async<6>();     // current tile's 6 copies done (in-order)
        } else {
            wait_async<0>();
        }
        __syncthreads();         // all waves' copies for buf[cur] visible

        v16h af0 = build_a_frag(Ast[cur], wave * 32 + r, hi);
        v16h af1 = build_a_frag(Ast[cur], wave * 32 + 16 + r, hi);
        #pragma unroll
        for (int nt = 0; nt < 4; ++nt) {
            v16h bf = build_b_frag(Bt[cur], nt * 16 + r, hi);
            acc[0][nt] = __builtin_amdgcn_wmma_f32_16x16x32_f16(
                false, af0, false, bf, (short)0, acc[0][nt], false, false);
            acc[1][nt] = __builtin_amdgcn_wmma_f32_16x16x32_f16(
                false, af1, false, bf, (short)0, acc[1][nt], false, false);
        }
        __syncthreads();         // buf[cur] fully consumed before re-staged
    }

    if constexpr (PASS == 1) {
        _Float16* tT = (_Float16*)dstp;
        #pragma unroll
        for (int rg = 0; rg < 2; ++rg) {
            const int rowb = m0 + wave * 32 + rg * 16 + hi * 8;
            float iv[8];
            #pragma unroll
            for (int c = 0; c < 8; ++c) iv[c] = inv[(size_t)bh * SEQ_L + rowb + c];
            #pragma unroll
            for (int nt = 0; nt < 4; ++nt) {
                union { uint4 u; _Float16 hh[8]; } pk;
                #pragma unroll
                for (int c = 0; c < 8; ++c)
                    pk.hh[c] = (_Float16)(acc[rg][nt][c] * iv[c]);
                // tT[bh][d = nt*16+r][m = rowb..rowb+7]  (contiguous 16B store)
                *(uint4*)(tT + ((size_t)bh * EDIM + nt * 16 + r) * SEQ_L + rowb) = pk.u;
            }
        }
    } else {
        float* out = (float*)dstp;
        const int b = bh >> 3, h = bh & 7;
        #pragma unroll
        for (int rg = 0; rg < 2; ++rg) {
            const int rowb = m0 + wave * 32 + rg * 16 + hi * 8;
            #pragma unroll
            for (int c = 0; c < 8; ++c) {
                size_t base = ((size_t)(b * SEQ_L + rowb + c) * NHEADS + h) * EDIM;
                #pragma unroll
                for (int nt = 0; nt < 4; ++nt)
                    out[base + nt * 16 + r] = acc[rg][nt][c];
            }
        }
    }
}

// ---------------------------------------------------------------------------
// Host launcher. Inputs: values, conv_w, conv_b, U, S (all fp32).
// Workspace: U16 @0 (512K) | UT16 @512K (512K) | vT16 @1M (8M) |
//            inv @9437184 (256K) | tT16 @9699328 (8M)   (~18.1 MB total)
// ---------------------------------------------------------------------------
extern "C" void kernel_launch(void* const* d_in, const int* in_sizes, int n_in,
                              void* d_out, int out_size, void* d_ws, size_t ws_size,
                              hipStream_t stream) {
    const float* values = (const float*)d_in[0];
    const float* conv_w = (const float*)d_in[1];
    const float* conv_b = (const float*)d_in[2];
    const float* U      = (const float*)d_in[3];
    const float* S      = (const float*)d_in[4];
    float* out = (float*)d_out;

    _Float16* U16  = (_Float16*)d_ws;
    _Float16* UT16 = (_Float16*)((char*)d_ws + 524288);
    _Float16* vT16 = (_Float16*)((char*)d_ws + 1048576);
    float*    inv  = (float*)((char*)d_ws + 9437184);
    _Float16* tT16 = (_Float16*)((char*)d_ws + 9699328);

    prep_u_kernel<<<dim3(64), dim3(256), 0, stream>>>(U, U16, UT16);
    prep_v_kernel<<<dim3(8, BH), dim3(256), 0, stream>>>(values, vT16);
    lambda_inv_kernel<<<dim3(BH * SEQ_L / 256), dim3(256), 0, stream>>>(
        values, conv_w, conv_b, S, inv);

    dim3 gg(SEQ_L / MTILE, BH);   // (4, 128)
    gemm_kernel<1><<<gg, dim3(128), 0, stream>>>(UT16, vT16, inv, (void*)tT16);
    gemm_kernel<2><<<gg, dim3(128), 0, stream>>>(U16, tT16, inv, (void*)out);
}
